// MultiHeadLatentAttention_75831942578277
// MI455X (gfx1250) — matrix operand order repaired and checked by reading, not compiled, not run on hardware
//
#include <hip/hip_runtime.h>

// ---------- problem constants ----------
#define TT 4096   // sequence length
#define CC 1024   // embed dim
#define HH 16     // heads
#define DD 64     // head dim
#define N3 3072   // 3*CC

typedef __attribute__((ext_vector_type(16))) __bf16 v16bf;
typedef __attribute__((ext_vector_type(8)))  float  v8f;

union FragAB {
    v16bf v;
    uint4 q[2];
};

static __device__ __forceinline__ v8f wmma_bf16(const FragAB& a, const FragAB& b, v8f c) {
    return __builtin_amdgcn_wmma_f32_16x16x32_bf16(
        /*neg_a=*/false, a.v, /*neg_b=*/false, b.v,
        /*c_mod=*/(short)0, c, /*reuse_a=*/false, /*reuse_b=*/false);
}

static __device__ __forceinline__ unsigned short f2bf(float f) {
    unsigned int u = __float_as_uint(f);
    u += 0x7fffu + ((u >> 16) & 1u);   // round-to-nearest-even
    return (unsigned short)(u >> 16);
}

// A-operand (16x32 bf16, row-major src): lane holds row m=lane%16,
// halves e=0..7 -> k = grp*8+e ; e=8..15 -> k = grp*8+16+(e-8)
static __device__ __forceinline__ void load_fragA(FragAB& f, const unsigned short* p) {
    f.q[0] = *reinterpret_cast<const uint4*>(p);        // k = base .. base+7
    f.q[1] = *reinterpret_cast<const uint4*>(p + 16);   // k = base+16 .. base+23
}
// B-operand (32x16 bf16, stored [N][K]): lane holds col n=lane%16,
// halves e=0..7 -> k = grp*8+e ; e=8..15 -> k = grp*8+16+(e-8)
static __device__ __forceinline__ void load_fragB(FragAB& f, const unsigned short* p) {
    f.q[0] = *reinterpret_cast<const uint4*>(p);
    f.q[1] = *reinterpret_cast<const uint4*>(p + 16);
}

static __device__ __forceinline__ float redmax16(float v) {
    v = fmaxf(v, __shfl_xor(v, 1, 32));
    v = fmaxf(v, __shfl_xor(v, 2, 32));
    v = fmaxf(v, __shfl_xor(v, 4, 32));
    v = fmaxf(v, __shfl_xor(v, 8, 32));
    return v;
}
static __device__ __forceinline__ float redsum16(float v) {
    v += __shfl_xor(v, 1, 32);
    v += __shfl_xor(v, 2, 32);
    v += __shfl_xor(v, 4, 32);
    v += __shfl_xor(v, 8, 32);
    return v;
}

// ---------- prep kernels ----------
// vectorized f32 -> bf16 convert, 4 elements/thread
__global__ void k_cvt_bf16(const float* __restrict__ src, unsigned short* __restrict__ dst, int n4) {
    int i = blockIdx.x * blockDim.x + threadIdx.x;
    if (i < n4) {
        float4 f = reinterpret_cast<const float4*>(src)[i];
        uint2 p;
        p.x = (unsigned int)f2bf(f.x) | ((unsigned int)f2bf(f.y) << 16);
        p.y = (unsigned int)f2bf(f.z) | ((unsigned int)f2bf(f.w) << 16);
        reinterpret_cast<uint2*>(dst)[i] = p;
    }
}

// w[K][N] (row-major) -> wt[N][K] bf16, 32x32 tile through LDS (coalesced both ways)
__global__ void k_transpose_bf16(const float* __restrict__ w, unsigned short* __restrict__ wt,
                                 int K, int N) {
    __shared__ float tile[32][33];
    int tx = threadIdx.x & 31, ty = threadIdx.x >> 5;     // 32 x 8
    int n0 = blockIdx.x * 32, k0 = blockIdx.y * 32;
#pragma unroll
    for (int j = 0; j < 4; ++j)
        tile[ty + j * 8][tx] = w[(k0 + ty + j * 8) * N + n0 + tx];
    __syncthreads();
#pragma unroll
    for (int j = 0; j < 4; ++j)
        wt[(n0 + ty + j * 8) * K + k0 + tx] = f2bf(tile[tx][ty + j * 8]);
}

// ---------- GEMM helpers: 32(M) x 64(N) per wave, double-buffered fragments ----------
static __device__ __forceinline__ void loadA2(FragAB a[2], const unsigned short* __restrict__ base,
                                              int row0, int ld, int k0, int n16, int grp) {
#pragma unroll
    for (int mt = 0; mt < 2; ++mt)
        load_fragA(a[mt], base + (row0 + mt * 16 + n16) * ld + k0 + grp * 8);
}
static __device__ __forceinline__ void loadB4(FragAB b[4], const unsigned short* __restrict__ base,
                                              int col0, int ld, int k0, int n16, int grp) {
#pragma unroll
    for (int nt = 0; nt < 4; ++nt)
        load_fragB(b[nt], base + (col0 + nt * 16 + n16) * ld + k0 + grp * 8);
}
static __device__ __forceinline__ void mma8(v8f acc[2][4], const FragAB a[2], const FragAB b[4]) {
#pragma unroll
    for (int mt = 0; mt < 2; ++mt)
#pragma unroll
        for (int nt = 0; nt < 4; ++nt)
            acc[mt][nt] = wmma_bf16(a[mt], b[nt], acc[mt][nt]);
}

// ---------- QKV projection: [TT,CC] @ [CC,N3] -> Q,K (row-major per head), V (transposed) ----------
__global__ void k_qkv_gemm(const unsigned short* __restrict__ Xb,
                           const unsigned short* __restrict__ Wt,   // [N3][CC]
                           unsigned short* __restrict__ Qb,         // [H][T][D], pre-scaled
                           unsigned short* __restrict__ Kb,         // [H][T][D]
                           unsigned short* __restrict__ Vt) {       // [H][D][T]
    const int NSTRIP = N3 / 64;                       // 48
    int wid  = (blockIdx.x * blockDim.x + threadIdx.x) >> 5;
    int lane = threadIdx.x & 31;
    int grp  = lane >> 4, n16 = lane & 15;
    int mtile = wid / NSTRIP, s = wid - mtile * NSTRIP;
    int row0 = mtile * 32, col0 = s * 64;

    v8f acc[2][4] = {};
    FragAB a0[2], b0[4], a1[2], b1[4];
    loadA2(a0, Xb, row0, CC, 0, n16, grp);
    loadB4(b0, Wt, col0, CC, 0, n16, grp);
    for (int k0 = 0; k0 < CC; k0 += 64) {
        loadA2(a1, Xb, row0, CC, k0 + 32, n16, grp);     // prefetch stage 1
        loadB4(b1, Wt, col0, CC, k0 + 32, n16, grp);
        mma8(acc, a0, b0);                               // compute stage 0
        if (k0 + 64 < CC) {
            loadA2(a0, Xb, row0, CC, k0 + 64, n16, grp); // prefetch stage 0'
            loadB4(b0, Wt, col0, CC, k0 + 64, n16, grp);
        }
        mma8(acc, a1, b1);                               // compute stage 1
    }

    // uniform routing: a 64-wide strip never crosses a Q/K/V boundary
    int seg = col0 >> 10;              // 0=Q, 1=K, 2=V
    int h   = (col0 & (CC - 1)) >> 6;  // head index for this strip
    if (seg == 0) {
#pragma unroll
        for (int mt = 0; mt < 2; ++mt)
#pragma unroll
            for (int nt = 0; nt < 4; ++nt) {
                int d = nt * 16 + n16;
#pragma unroll
                for (int r = 0; r < 8; ++r) {
                    int row = row0 + mt * 16 + grp * 8 + r;
                    Qb[(h * TT + row) * DD + d] = f2bf(acc[mt][nt][r] * 0.125f);
                }
            }
    } else if (seg == 1) {
#pragma unroll
        for (int mt = 0; mt < 2; ++mt)
#pragma unroll
            for (int nt = 0; nt < 4; ++nt) {
                int d = nt * 16 + n16;
#pragma unroll
                for (int r = 0; r < 8; ++r) {
                    int row = row0 + mt * 16 + grp * 8 + r;
                    Kb[(h * TT + row) * DD + d] = f2bf(acc[mt][nt][r]);
                }
            }
    } else {
#pragma unroll
        for (int mt = 0; mt < 2; ++mt)
#pragma unroll
            for (int nt = 0; nt < 4; ++nt) {
                int d = nt * 16 + n16;
#pragma unroll
                for (int r = 0; r < 8; ++r) {
                    int row = row0 + mt * 16 + grp * 8 + r;
                    Vt[(h * DD + d) * TT + row] = f2bf(acc[mt][nt][r]);
                }
            }
    }
}

// ---------- flash attention: 2 query tiles (32 rows) per wave ----------
// one 32-key chunk for both q-tiles: 8 S-wmmas on prefetched K frags, prefetch
// V frags (this chunk) + K frags (next chunk) before the VALU-heavy softmax so
// vmem latency overlaps exp/shuffle work; every K/V fragment is reused x2.
static __device__ __forceinline__ void attn_step(
    int kc, int kcn, int h, int n16, int grp,
    const FragAB qf[2][2], const FragAB kf[4], FragAB kfn[4],
    const unsigned short* __restrict__ Kb,
    const unsigned short* __restrict__ Vt,
    float mi[2][8], float li[2][8], v8f o[2][4],
    unsigned short* pl) {                      // 2 P tiles, stride 512 ushorts
    // scores: 2 x (16 x 32)
    v8f s[2][2] = {};
#pragma unroll
    for (int qt = 0; qt < 2; ++qt)
#pragma unroll
        for (int nt = 0; nt < 2; ++nt)
#pragma unroll
            for (int seg = 0; seg < 2; ++seg)
                s[qt][nt] = wmma_bf16(qf[qt][seg], kf[nt * 2 + seg], s[qt][nt]);

    // V fragments for THIS chunk (independent of softmax -> issue early)
    FragAB vf[4];
#pragma unroll
    for (int nt = 0; nt < 4; ++nt)
        load_fragB(vf[nt], Vt + (h * DD + nt * 16 + n16) * TT + kc + grp * 8);

    // K fragments for NEXT chunk (double-buffer)
    if (kcn < TT) {
#pragma unroll
        for (int nt = 0; nt < 2; ++nt)
#pragma unroll
            for (int seg = 0; seg < 2; ++seg)
                load_fragB(kfn[nt * 2 + seg],
                           Kb + (h * TT + kcn + nt * 16 + n16) * DD + seg * 32 + grp * 8);
    }

    // online softmax (rows at VGPR index r, 16 cols across the lane group)
#pragma unroll
    for (int qt = 0; qt < 2; ++qt) {
#pragma unroll
        for (int r = 0; r < 8; ++r) {
            float v = fmaxf(s[qt][0][r], s[qt][1][r]);
            v = redmax16(v);
            float mnew  = fmaxf(mi[qt][r], v);
            float alpha = __expf(mi[qt][r] - mnew);
            float p0 = __expf(s[qt][0][r] - mnew);
            float p1 = __expf(s[qt][1][r] - mnew);
            s[qt][0][r] = p0; s[qt][1][r] = p1;
            float rs = redsum16(p0 + p1);
            li[qt][r] = li[qt][r] * alpha + rs;
            mi[qt][r] = mnew;
#pragma unroll
            for (int nt = 0; nt < 4; ++nt) o[qt][nt][r] *= alpha;
        }
    }

    // P: C-layout -> LDS -> A-layout fragments
#pragma unroll
    for (int qt = 0; qt < 2; ++qt)
#pragma unroll
        for (int nt = 0; nt < 2; ++nt)
#pragma unroll
            for (int r = 0; r < 8; ++r)
                pl[qt * 512 + (grp * 8 + r) * 32 + nt * 16 + n16] = f2bf(s[qt][nt][r]);
    FragAB pf[2];
#pragma unroll
    for (int qt = 0; qt < 2; ++qt)
        load_fragA(pf[qt], pl + qt * 512 + n16 * 32 + grp * 8);

    // O += P(16x32) @ V(32x64), V fragments reused for both q-tiles
#pragma unroll
    for (int qt = 0; qt < 2; ++qt)
#pragma unroll
        for (int nt = 0; nt < 4; ++nt)
            o[qt][nt] = wmma_bf16(pf[qt], vf[nt], o[qt][nt]);
}

__global__ void k_attn(const unsigned short* __restrict__ Qb,
                       const unsigned short* __restrict__ Kb,
                       const unsigned short* __restrict__ Vt,
                       unsigned short* __restrict__ Yb) {           // [T][C] bf16
    __shared__ unsigned short plds[8][2][16][32];                   // 2 P tiles per wave
    int w    = threadIdx.x >> 5;
    int lane = threadIdx.x & 31;
    int grp  = lane >> 4, n16 = lane & 15;
    int wid  = blockIdx.x * 8 + w;
    int h   = wid >> 7;                // 128 double-q-tiles per head
    int qt2 = wid & 127;
    int q0  = qt2 * 32;
    unsigned short* pl = &plds[w][0][0][0];

    // Q fragments: 2 tiles x (16 x 64), two K=32 segments each
    FragAB qf[2][2];
#pragma unroll
    for (int qt = 0; qt < 2; ++qt) {
        const unsigned short* pq = Qb + (h * TT + q0 + qt * 16 + n16) * DD + grp * 8;
        load_fragA(qf[qt][0], pq);
        load_fragA(qf[qt][1], pq + 32);
    }

    float mi[2][8], li[2][8];
#pragma unroll
    for (int qt = 0; qt < 2; ++qt)
#pragma unroll
        for (int r = 0; r < 8; ++r) { mi[qt][r] = -3.0e38f; li[qt][r] = 0.0f; }
    v8f o[2][4] = {};

    // preload K frags for first chunk, then ping-pong two chunks per iteration
    FragAB kf0[4], kf1[4];
#pragma unroll
    for (int nt = 0; nt < 2; ++nt)
#pragma unroll
        for (int seg = 0; seg < 2; ++seg)
            load_fragB(kf0[nt * 2 + seg],
                       Kb + (h * TT + nt * 16 + n16) * DD + seg * 32 + grp * 8);

    for (int kc = 0; kc < TT; kc += 64) {
        attn_step(kc,      kc + 32, h, n16, grp, qf, kf0, kf1, Kb, Vt, mi, li, o, pl);
        attn_step(kc + 32, kc + 64, h, n16, grp, qf, kf1, kf0, Kb, Vt, mi, li, o, pl);
    }

    // epilogue: normalize, scatter into [T][C] with head offset
#pragma unroll
    for (int qt = 0; qt < 2; ++qt)
#pragma unroll
        for (int nt = 0; nt < 4; ++nt) {
            int col = h * DD + nt * 16 + n16;
#pragma unroll
            for (int r = 0; r < 8; ++r) {
                int row = q0 + qt * 16 + grp * 8 + r;
                Yb[row * CC + col] = f2bf(o[qt][nt][r] / li[qt][r]);
            }
        }
}

// ---------- output projection: [TT,CC] @ [CC,CC] -> fp32 out ----------
__global__ void k_out_gemm(const unsigned short* __restrict__ Yb,
                           const unsigned short* __restrict__ Wt,   // [CC][CC] = WoutT
                           float* __restrict__ out) {
    const int NSTRIP = CC / 64;                       // 16
    int wid  = (blockIdx.x * blockDim.x + threadIdx.x) >> 5;
    int lane = threadIdx.x & 31;
    int grp  = lane >> 4, n16 = lane & 15;
    int mtile = wid / NSTRIP, s = wid - mtile * NSTRIP;
    int row0 = mtile * 32, col0 = s * 64;

    v8f acc[2][4] = {};
    FragAB a0[2], b0[4], a1[2], b1[4];
    loadA2(a0, Yb, row0, CC, 0, n16, grp);
    loadB4(b0, Wt, col0, CC, 0, n16, grp);
    for (int k0 = 0; k0 < CC; k0 += 64) {
        loadA2(a1, Yb, row0, CC, k0 + 32, n16, grp);
        loadB4(b1, Wt, col0, CC, k0 + 32, n16, grp);
        mma8(acc, a0, b0);
        if (k0 + 64 < CC) {
            loadA2(a0, Yb, row0, CC, k0 + 64, n16, grp);
            loadB4(b0, Wt, col0, CC, k0 + 64, n16, grp);
        }
        mma8(acc, a1, b1);
    }
#pragma unroll
    for (int mt = 0; mt < 2; ++mt)
#pragma unroll
        for (int nt = 0; nt < 4; ++nt) {
            int col = col0 + nt * 16 + n16;
#pragma unroll
            for (int r = 0; r < 8; ++r) {
                int row = row0 + mt * 16 + grp * 8 + r;
                out[row * CC + col] = acc[mt][nt][r];
            }
        }
}

// ---------- launch ----------
extern "C" void kernel_launch(void* const* d_in, const int* in_sizes, int n_in,
                              void* d_out, int out_size, void* d_ws, size_t ws_size,
                              hipStream_t stream) {
    const float* x     = (const float*)d_in[0];   // [1,T,C]
    const float* w_qkv = (const float*)d_in[1];   // [C,3C]
    const float* w_out = (const float*)d_in[2];   // [C,C]
    float* out = (float*)d_out;

    char* ws = (char*)d_ws;
    // workspace layout (bf16 = 2 bytes), 48 MB total
    unsigned short* Xb    = (unsigned short*)(ws + 0);              //  8 MB [T][C]
    unsigned short* Wqkvt = (unsigned short*)(ws + (8u  << 20));    //  6 MB [3C][C]
    unsigned short* Woutt = (unsigned short*)(ws + (14u << 20));    //  2 MB [C][C]
    unsigned short* Qb    = (unsigned short*)(ws + (16u << 20));    //  8 MB [H][T][D]
    unsigned short* Kb    = (unsigned short*)(ws + (24u << 20));    //  8 MB [H][T][D]
    unsigned short* Vt    = (unsigned short*)(ws + (32u << 20));    //  8 MB [H][D][T]
    unsigned short* Yb    = (unsigned short*)(ws + (40u << 20));    //  8 MB [T][C]

    // 1) convert activations / weights to bf16 (weights transposed to [N][K])
    k_cvt_bf16<<<(TT * CC / 4 + 255) / 256, 256, 0, stream>>>(x, Xb, TT * CC / 4);
    k_transpose_bf16<<<dim3(N3 / 32, CC / 32), 256, 0, stream>>>(w_qkv, Wqkvt, CC, N3);
    k_transpose_bf16<<<dim3(CC / 32, CC / 32), 256, 0, stream>>>(w_out, Woutt, CC, CC);

    // 2) QKV projection: waves = (T/32)*(3C/64) = 6144 -> 768 blocks of 8 waves
    k_qkv_gemm<<<768, 256, 0, stream>>>(Xb, Wqkvt, Qb, Kb, Vt);

    // 3) flash attention: waves = H*(T/32) = 2048 -> 256 blocks
    k_attn<<<256, 256, 0, stream>>>(Qb, Kb, Vt, Yb);

    // 4) output projection: waves = (T/32)*(C/64) = 2048 -> 256 blocks
    k_out_gemm<<<256, 256, 0, stream>>>(Yb, Woutt, out);
}
